// GMNnet_15839839387945
// MI455X (gfx1250) — matrix operand order repaired and checked by reading, not compiled, not run on hardware
//
#include <hip/hip_runtime.h>
#include <hip/hip_bf16.h>

// ---------------------------------------------------------------------------
// Types for CDNA5 WMMA (wave32, 16x16x32 bf16 -> f32)
// ---------------------------------------------------------------------------
typedef __bf16 bf16;
typedef __attribute__((ext_vector_type(16))) __bf16 v16bf;
typedef __attribute__((ext_vector_type(8)))  __bf16 v8bf;
typedef __attribute__((ext_vector_type(8)))  float  v8f;

union BF16x16 { v16bf v; v8bf h[2]; __bf16 e[16]; };
union F8      { v8f v; float f[8]; };

#define DIM 128

// A-operand loader: row-major MxK tile (stride in elements), 16x32 bf16 slab.
// Contiguous k order, same as load_Bt: lane (m = lane&15, hh = lane>>4) holds K = hh*16+[0..15].
__device__ __forceinline__ v16bf load_A32(const bf16* base, int strideElems) {
  int lane = threadIdx.x & 31;
  int m = lane & 15, hh = lane >> 4;
  BF16x16 u;
  u.h[0] = *(const v8bf*)(base + m * strideElems + hh * 16);
  u.h[1] = *(const v8bf*)(base + m * strideElems + hh * 16 + 8);
  return u.v;
}

// B-operand loader from B^T stored row-major (N x K, stride in elements).
// Contiguous k order, same as load_A32: lane (n = lane&15, hh = lane>>4) holds K = hh*16+[0..15] of column n.
__device__ __forceinline__ v16bf load_Bt(const bf16* baseT, int strideElems) {
  int lane = threadIdx.x & 31;
  int n = lane & 15, hh = lane >> 4;
  return *(const v16bf*)(baseT + n * strideElems + hh * 16);
}

__device__ __forceinline__ v8f wmma_bf16(v16bf a, v16bf b, v8f c) {
  return __builtin_amdgcn_wmma_f32_16x16x32_bf16(false, a, false, b, (short)0, c,
                                                 false, false);
}

// ---------------------------------------------------------------------------
// Small prep kernels
// ---------------------------------------------------------------------------
__global__ void zero_f32(float* p, int n) {
  int t = blockIdx.x * blockDim.x + threadIdx.x;
  if (t < n) p[t] = 0.f;
}

// dst (bf16, N x K row-major) = transpose of src (f32, K x N row-major)
__global__ void transpose_to_bf16(const float* __restrict__ src,
                                  bf16* __restrict__ dst, int K, int N) {
  int t = blockIdx.x * blockDim.x + threadIdx.x;
  if (t >= K * N) return;
  int k = t / N, n = t % N;
  dst[n * K + k] = (bf16)src[k * N + n];
}

__global__ void gather_embed(const float* __restrict__ W, const int* __restrict__ idx,
                             float* __restrict__ xf, bf16* __restrict__ xb, int N) {
  int t = blockIdx.x * blockDim.x + threadIdx.x;
  if (t >= N * DIM) return;
  int i = t >> 7, f = t & 127;
  float v = W[idx[i] * DIM + f];
  xf[t] = v;
  xb[t] = (bf16)v;
}

__global__ void gather_edge_embed(const float* __restrict__ W, const int* __restrict__ attr,
                                  bf16* __restrict__ ewb, int E) {
  int t = blockIdx.x * blockDim.x + threadIdx.x;
  if (t >= E * DIM) return;
  int e = t >> 7, f = t & 127;
  ewb[t] = (bf16)W[attr[e] * DIM + f];
}

// ---------------------------------------------------------------------------
// Message passing: per-edge relu([x[dst]|x[src]|ew] @ fmsg_W + b), atomically
// scattered into msg[dst]. WMMA-tiled: 16 edges x 128 out, K=384 (12 chunks).
// ---------------------------------------------------------------------------
__global__ void __launch_bounds__(128) propagate_kernel(
    const bf16* __restrict__ xb, const bf16* __restrict__ ewb,
    const int* __restrict__ srcI, const int* __restrict__ dstI,
    const bf16* __restrict__ WT,       // (128 x 384) bf16 = fmsg_W^T row-major
    const float* __restrict__ bias, float* __restrict__ msg, int E) {
  __shared__ __align__(32) bf16 ldsF[4][16 * 384];
  int tid = threadIdx.x, wave = tid >> 5, lane = tid & 31;
  int e0 = blockIdx.x * 64 + wave * 16;
  if (e0 >= E) return;  // per-wave LDS only; no block barriers in this kernel

  for (int idx = lane; idx < 16 * 384; idx += 32) {
    int e = idx / 384, k = idx - e * 384;
    int ge = e0 + e;
    bf16 v;
    if (k < 128)        v = xb[dstI[ge] * DIM + k];
    else if (k < 256)   v = xb[srcI[ge] * DIM + (k - 128)];
    else                v = ewb[ge * DIM + (k - 256)];
    ldsF[wave][idx] = v;
  }

  int n = lane & 15, hh = lane >> 4;
  int drow[8];
  for (int r = 0; r < 8; ++r) drow[r] = dstI[e0 + hh * 8 + r];

  for (int nt = 0; nt < 8; ++nt) {
    F8 a;
    for (int i = 0; i < 8; ++i) a.f[i] = 0.f;
    for (int k = 0; k < 12; ++k) {
      v16bf av = load_A32(&ldsF[wave][k * 32], 384);
      v16bf bv = load_Bt(WT + (nt * 16) * 384 + k * 32, 384);
      a.v = wmma_bf16(av, bv, a.v);
    }
    int col = nt * 16 + n;
    float b = bias[col];
    for (int r = 0; r < 8; ++r) {
      float v = a.f[r] + b;
      if (v > 0.f) atomicAdd(&msg[drow[r] * DIM + col], v);
    }
  }
}

// ---------------------------------------------------------------------------
// Fused cross-attention: U = Qf - softmax_row(Qb @ Kb^T) @ Kb  (flash style).
// 256 threads = 8 waves; each wave owns a 16-row M tile; block shares 32-key
// KV tiles staged in LDS (row-major for S, transposed for P@V).
// ---------------------------------------------------------------------------
__global__ void __launch_bounds__(256) flash_cross_attn(
    const bf16* __restrict__ Qb, const bf16* __restrict__ Kb,
    const float* __restrict__ Qf, float* __restrict__ U, int Nq, int Nk) {
  __shared__ __align__(32) bf16 ldsK[32 * DIM];   // keys row-major
  __shared__ __align__(32) bf16 ldsVt[DIM * 32];  // values transposed (feat x key)
  __shared__ __align__(32) bf16 ldsP[8][16 * 32]; // per-wave P staging

  const int tid = threadIdx.x, wave = tid >> 5, lane = tid & 31;
  const int n = lane & 15, hh = lane >> 4;
  const int m0 = blockIdx.x * 128 + wave * 16;
  const bool active = (m0 < Nq);

  v16bf aQ[4];
  if (active)
    for (int k = 0; k < 4; ++k) aQ[k] = load_A32(Qb + m0 * DIM + k * 32, DIM);

  F8 acc[8];
  float mrow[8], lrow[8];
  for (int nt = 0; nt < 8; ++nt)
    for (int i = 0; i < 8; ++i) acc[nt].f[i] = 0.f;
  for (int r = 0; r < 8; ++r) { mrow[r] = -3.0e38f; lrow[r] = 0.f; }

  for (int kv0 = 0; kv0 < Nk; kv0 += 32) {
    __syncthreads();
    {  // stage KV tile: 32 keys x 128 feats, row-major + transposed
      int key = tid >> 3;
      int fb = (tid & 7) * 16;
      BF16x16 val;
      if (kv0 + key < Nk) val.v = *(const v16bf*)(Kb + (kv0 + key) * DIM + fb);
      else for (int i = 0; i < 16; ++i) val.e[i] = (bf16)0.f;
      *(v16bf*)(&ldsK[key * DIM + fb]) = val.v;
      for (int i = 0; i < 16; ++i) ldsVt[(fb + i) * 32 + key] = val.e[i];
      if (kv0 + 32 < Nk) __builtin_prefetch(Kb + (kv0 + 32 + key) * DIM, 0, 0);
    }
    __syncthreads();

    if (active) {
      for (int c = 0; c < 2; ++c) {
        F8 s;
        for (int i = 0; i < 8; ++i) s.f[i] = 0.f;
        for (int k = 0; k < 4; ++k) {
          v16bf bv = load_Bt(&ldsK[(c * 16) * DIM + k * 32], DIM);
          s.v = wmma_bf16(aQ[k], bv, s.v);
        }
        bool kvalid = (kv0 + c * 16 + n) < Nk;
        for (int r = 0; r < 8; ++r) {
          float v = s.f[r];
          v = fmaxf(v, __shfl_xor(v, 1));
          v = fmaxf(v, __shfl_xor(v, 2));
          v = fmaxf(v, __shfl_xor(v, 4));
          v = fmaxf(v, __shfl_xor(v, 8));
          float mn = fmaxf(mrow[r], v);
          float alpha = __expf(mrow[r] - mn);
          mrow[r] = mn;
          float p = kvalid ? __expf(s.f[r] - mn) : 0.f;
          float ps = p;
          ps += __shfl_xor(ps, 1);
          ps += __shfl_xor(ps, 2);
          ps += __shfl_xor(ps, 4);
          ps += __shfl_xor(ps, 8);
          lrow[r] = lrow[r] * alpha + ps;
          for (int nt = 0; nt < 8; ++nt) acc[nt].f[r] *= alpha;
          ldsP[wave][(hh * 8 + r) * 32 + c * 16 + n] = (bf16)p;
        }
      }
      v16bf aP = load_A32(ldsP[wave], 32);  // same-wave LDS RAW: in-order
      for (int nt = 0; nt < 8; ++nt) {
        v16bf bv = load_Bt(&ldsVt[(nt * 16) * 32], 32);
        acc[nt].v = wmma_bf16(aP, bv, acc[nt].v);
      }
    }
  }

  if (active) {
    for (int nt = 0; nt < 8; ++nt)
      for (int r = 0; r < 8; ++r) {
        int row = m0 + hh * 8 + r, col = nt * 16 + n;
        float o = acc[nt].f[r] / lrow[r];
        U[row * DIM + col] = Qf[row * DIM + col] - o;
      }
  }
}

// ---------------------------------------------------------------------------
// GRU update: tiles of 16 nodes; g_i = [m|u]@Wih, g_h = h@Whh via WMMA,
// gate nonlinearity in f32, writes new h (f32 + bf16 mirror).
// ---------------------------------------------------------------------------
__global__ void __launch_bounds__(128) gru_kernel(
    const float* __restrict__ mf, const float* __restrict__ uf,
    const float* __restrict__ hf,
    const bf16* __restrict__ WihT,  // (384 x 256)
    const bf16* __restrict__ WhhT,  // (384 x 128)
    const float* __restrict__ bih, const float* __restrict__ bhh,
    float* __restrict__ outf, bf16* __restrict__ outb, int N) {
  __shared__ __align__(32) bf16 ldsA1[4][16 * 256];
  __shared__ __align__(32) bf16 ldsA2[4][16 * 128];
  int tid = threadIdx.x, wave = tid >> 5, lane = tid & 31;
  int n0 = blockIdx.x * 64 + wave * 16;
  if (n0 >= N) return;

  for (int idx = lane; idx < 16 * 256; idx += 32) {
    int e = idx >> 8, k = idx & 255;
    float v = (k < 128) ? mf[(n0 + e) * DIM + k] : uf[(n0 + e) * DIM + (k - 128)];
    ldsA1[wave][idx] = (bf16)v;
  }
  for (int idx = lane; idx < 16 * 128; idx += 32) {
    int e = idx >> 7, k = idx & 127;
    ldsA2[wave][idx] = (bf16)hf[(n0 + e) * DIM + k];
  }

  int nn = lane & 15, hh = lane >> 4;
  for (int j = 0; j < 8; ++j) {
    F8 gi[3], gh[3];
    for (int g = 0; g < 3; ++g)
      for (int i = 0; i < 8; ++i) { gi[g].f[i] = 0.f; gh[g].f[i] = 0.f; }
    for (int g = 0; g < 3; ++g) {
      const bf16* Bi = WihT + (g * 128 + j * 16) * 256;
      for (int k = 0; k < 8; ++k)
        gi[g].v = wmma_bf16(load_A32(&ldsA1[wave][k * 32], 256),
                            load_Bt(Bi + k * 32, 256), gi[g].v);
      const bf16* Bh = WhhT + (g * 128 + j * 16) * 128;
      for (int k = 0; k < 4; ++k)
        gh[g].v = wmma_bf16(load_A32(&ldsA2[wave][k * 32], 128),
                            load_Bt(Bh + k * 32, 128), gh[g].v);
    }
    int col = j * 16 + nn;
    float bir = bih[col], biz = bih[128 + col], bin = bih[256 + col];
    float bhr = bhh[col], bhz = bhh[128 + col], bhn = bhh[256 + col];
    for (int r = 0; r < 8; ++r) {
      int row = n0 + hh * 8 + r;
      float hv = hf[row * DIM + col];
      float rr = 1.f / (1.f + __expf(-(gi[0].f[r] + bir + gh[0].f[r] + bhr)));
      float zz = 1.f / (1.f + __expf(-(gi[1].f[r] + biz + gh[1].f[r] + bhz)));
      float nv = tanhf(gi[2].f[r] + bin + rr * (gh[2].f[r] + bhn));
      float nh = (1.f - zz) * nv + zz * hv;
      outf[row * DIM + col] = nh;
      outb[row * DIM + col] = (bf16)nh;
    }
  }
}

// ---------------------------------------------------------------------------
// Gated pooling: gate = softmax(sigmoid(x@gW+b)) over nodes; out = sum gate*x.
// Single 1024-thread block (tiny cost).
// ---------------------------------------------------------------------------
__global__ void __launch_bounds__(1024) pool_kernel(
    const float* __restrict__ x, const float* __restrict__ gW,
    const float* __restrict__ gb, float* __restrict__ out, int N) {
  __shared__ float t[10240];
  __shared__ float red[1024];
  __shared__ float partial[8 * 128];
  int tid = threadIdx.x;
  float lmax = -3e38f;
  for (int i = tid; i < N; i += 1024) {
    float d = 0.f;
    for (int f = 0; f < 128; ++f) d += x[i * DIM + f] * gW[f];
    d += gb[0];
    d = 1.f / (1.f + __expf(-d));  // sigmoid gate, then softmax over nodes
    t[i] = d;
    lmax = fmaxf(lmax, d);
  }
  red[tid] = lmax;
  __syncthreads();
  for (int s = 512; s > 0; s >>= 1) {
    if (tid < s) red[tid] = fmaxf(red[tid], red[tid + s]);
    __syncthreads();
  }
  float gmax = red[0];
  __syncthreads();
  float lsum = 0.f;
  for (int i = tid; i < N; i += 1024) {
    float e = __expf(t[i] - gmax);
    t[i] = e;
    lsum += e;
  }
  red[tid] = lsum;
  __syncthreads();
  for (int s = 512; s > 0; s >>= 1) {
    if (tid < s) red[tid] += red[tid + s];
    __syncthreads();
  }
  float gsum = red[0];
  __syncthreads();
  int f = tid & 127, sl = tid >> 7;
  float acc = 0.f;
  for (int i = sl; i < N; i += 8) acc += t[i] * x[i * DIM + f];
  partial[sl * 128 + f] = acc;
  __syncthreads();
  if (sl == 0) {
    float s0 = 0.f;
    for (int k = 0; k < 8; ++k) s0 += partial[k * 128 + f];
    out[f] = s0 / gsum;
  }
}

// ---------------------------------------------------------------------------
// Host orchestration
// ---------------------------------------------------------------------------
extern "C" void kernel_launch(void* const* d_in, const int* in_sizes, int n_in,
                              void* d_out, int out_size, void* d_ws, size_t ws_size,
                              hipStream_t stream) {
  const int*   x1_idx = (const int*)d_in[0];
  const int*   x2_idx = (const int*)d_in[1];
  const int*   ei1    = (const int*)d_in[2];
  const int*   ei2    = (const int*)d_in[3];
  const int*   ea1    = (const int*)d_in[4];
  const int*   ea2    = (const int*)d_in[5];
  const float* embedW = (const float*)d_in[6];
  const float* edgeW  = (const float*)d_in[7];
  const float* fmsgW  = (const float*)d_in[8];
  const float* fmsgB  = (const float*)d_in[9];
  const float* Wih    = (const float*)d_in[10];
  const float* bih    = (const float*)d_in[11];
  const float* Whh    = (const float*)d_in[12];
  const float* bhh    = (const float*)d_in[13];
  const float* gateW  = (const float*)d_in[14];
  const float* gateB  = (const float*)d_in[15];

  const int N1 = in_sizes[0], N2 = in_sizes[1];
  const int E1 = in_sizes[4], E2 = in_sizes[5];

  // workspace carving (256B aligned slices)
  char* p = (char*)d_ws;
  auto alloc = [&](size_t bytes) -> void* {
    void* r = (void*)p;
    p += (bytes + 255) & ~(size_t)255;
    return r;
  };
  float* x1f[2] = {(float*)alloc((size_t)N1 * DIM * 4), (float*)alloc((size_t)N1 * DIM * 4)};
  float* x2f[2] = {(float*)alloc((size_t)N2 * DIM * 4), (float*)alloc((size_t)N2 * DIM * 4)};
  bf16*  x1b[2] = {(bf16*)alloc((size_t)N1 * DIM * 2), (bf16*)alloc((size_t)N1 * DIM * 2)};
  bf16*  x2b[2] = {(bf16*)alloc((size_t)N2 * DIM * 2), (bf16*)alloc((size_t)N2 * DIM * 2)};
  bf16*  ew1b   = (bf16*)alloc((size_t)E1 * DIM * 2);
  bf16*  ew2b   = (bf16*)alloc((size_t)E2 * DIM * 2);
  float* msg1   = (float*)alloc((size_t)N1 * DIM * 4);
  float* msg2   = (float*)alloc((size_t)N2 * DIM * 4);
  float* u1     = (float*)alloc((size_t)N1 * DIM * 4);
  float* u2     = (float*)alloc((size_t)N2 * DIM * 4);
  bf16*  fmsgWT = (bf16*)alloc((size_t)128 * 384 * 2);
  bf16*  WihT   = (bf16*)alloc((size_t)384 * 256 * 2);
  bf16*  WhhT   = (bf16*)alloc((size_t)384 * 128 * 2);

  // weight prep + embeddings
  transpose_to_bf16<<<(384 * 128 + 255) / 256, 256, 0, stream>>>(fmsgW, fmsgWT, 384, 128);
  transpose_to_bf16<<<(256 * 384 + 255) / 256, 256, 0, stream>>>(Wih, WihT, 256, 384);
  transpose_to_bf16<<<(128 * 384 + 255) / 256, 256, 0, stream>>>(Whh, WhhT, 128, 384);
  gather_embed<<<(N1 * DIM + 255) / 256, 256, 0, stream>>>(embedW, x1_idx, x1f[0], x1b[0], N1);
  gather_embed<<<(N2 * DIM + 255) / 256, 256, 0, stream>>>(embedW, x2_idx, x2f[0], x2b[0], N2);
  gather_edge_embed<<<(E1 * DIM + 255) / 256, 256, 0, stream>>>(edgeW, ea1, ew1b, E1);
  gather_edge_embed<<<(E2 * DIM + 255) / 256, 256, 0, stream>>>(edgeW, ea2, ew2b, E2);

  int cur = 0;
  for (int layer = 0; layer < 2; ++layer) {
    zero_f32<<<(N1 * DIM + 255) / 256, 256, 0, stream>>>(msg1, N1 * DIM);
    zero_f32<<<(N2 * DIM + 255) / 256, 256, 0, stream>>>(msg2, N2 * DIM);
    propagate_kernel<<<(E1 + 63) / 64, 128, 0, stream>>>(
        x1b[cur], ew1b, ei1, ei1 + E1, fmsgWT, fmsgB, msg1, E1);
    propagate_kernel<<<(E2 + 63) / 64, 128, 0, stream>>>(
        x2b[cur], ew2b, ei2, ei2 + E2, fmsgWT, fmsgB, msg2, E2);
    flash_cross_attn<<<(N1 + 127) / 128, 256, 0, stream>>>(
        x1b[cur], x2b[cur], x1f[cur], u1, N1, N2);
    flash_cross_attn<<<(N2 + 127) / 128, 256, 0, stream>>>(
        x2b[cur], x1b[cur], x2f[cur], u2, N2, N1);
    gru_kernel<<<(N1 + 63) / 64, 128, 0, stream>>>(
        msg1, u1, x1f[cur], WihT, WhhT, bih, bhh, x1f[1 - cur], x1b[1 - cur], N1);
    gru_kernel<<<(N2 + 63) / 64, 128, 0, stream>>>(
        msg2, u2, x2f[cur], WihT, WhhT, bih, bhh, x2f[1 - cur], x2b[1 - cur], N2);
    cur ^= 1;
  }

  pool_kernel<<<1, 1024, 0, stream>>>(x1f[cur], gateW, gateB, (float*)d_out, N1);
  pool_kernel<<<1, 1024, 0, stream>>>(x2f[cur], gateW, gateB, (float*)d_out + DIM, N2);
}
